// GraphVectorQuantizer_24902220382506
// MI455X (gfx1250) — matrix operand order repaired and checked by reading, not compiled, not run on hardware
//
#include <hip/hip_runtime.h>
#include <cmath>

// ---------------------------------------------------------------------------
// Problem constants
// ---------------------------------------------------------------------------
#define BB 16
#define SS 1024
#define DIMM 512
#define NSYM 4096
#define PROJ 32
#define ROWS (BB * SS)            // 16384 (b,s) rows
#define TWOD (2 * DIMM)           // 1024
#define N_ELEM ((size_t)ROWS * DIMM) // 8388608 elements for variance

typedef __attribute__((ext_vector_type(16))) __bf16 v16bf;
typedef __attribute__((ext_vector_type(8)))  float  v8f;
typedef __attribute__((ext_vector_type(4)))  unsigned int u32x4;
typedef __attribute__((ext_vector_type(8)))  unsigned int u32x8;

// ---------------------------------------------------------------------------
// Helpers
// ---------------------------------------------------------------------------
__device__ __forceinline__ unsigned short f2bf(float x) {
  unsigned int u = __builtin_bit_cast(unsigned int, x);
  unsigned int r = u + 0x7FFFu + ((u >> 16) & 1u);
  return (unsigned short)(r >> 16);
}
__device__ __forceinline__ float bf2f(unsigned short h) {
  unsigned int u = ((unsigned int)h) << 16;
  return __builtin_bit_cast(float, u);
}
__device__ __forceinline__ float sigmoidf(float x) {
  return 1.0f / (1.0f + __expf(-x));
}

// Pack two 16-byte chunks into a 32-byte bf16x16 fragment.
__device__ __forceinline__ v16bf pack2(const unsigned short* p0,
                                       const unsigned short* p1) {
  u32x4 a = *(const u32x4*)p0;
  u32x4 b = *(const u32x4*)p1;
  u32x8 r;
  r[0] = a[0]; r[1] = a[1]; r[2] = a[2]; r[3] = a[3];
  r[4] = b[0]; r[5] = b[1]; r[6] = b[2]; r[7] = b[3];
  return __builtin_bit_cast(v16bf, r);
}
// A fragment (16x32 bf16): lane<16 -> M=lane, elems = K[k0..k0+7], K[k0+16..k0+23]
//                          lane>=16 -> M=lane-16, elems shifted by +8 in K.
__device__ __forceinline__ v16bf ldA(const unsigned short* row, int k0, int hi) {
  const unsigned short* p0 = row + k0 + hi * 8;
  return pack2(p0, p0 + 16);
}
// B fragment (32x16 bf16) from a K-major (transposed) matrix: row = column N,
// lane<16 holds K[k0..k0+7], K[k0+16..k0+23]; lane>=16 shifted by +8 in K.
__device__ __forceinline__ v16bf ldB(const unsigned short* row, int k0, int hi) {
  const unsigned short* p = row + k0 + hi * 8;
  return pack2(p, p + 16);
}
__device__ __forceinline__ v8f wmma_bf16(v16bf a, v16bf b, v8f c) {
  return __builtin_amdgcn_wmma_f32_16x16x32_bf16(
      false, a, false, b, (short)0, c, false, false);
}

// ---------------------------------------------------------------------------
// 4-stage ping-pong K-loops: four independent fragment register sets, each
// re-loaded fresh every iteration (no rotation copies -> no v_dual_mov storm;
// loop-carried values become stable PHI registers). Loads for k0+128/k0+160
// cross the backedge, keeping ~2 WMMA stages of L2 latency in flight.
// ---------------------------------------------------------------------------
template <int K>
__device__ __forceinline__ void gemm_pipe1(const unsigned short* arow,
                                           const unsigned short* brow, int hi,
                                           v8f& acc) {
  static_assert(K % 128 == 0 && K >= 256, "K layout");
  v16bf a0 = ldA(arow, 0, hi),  b0 = ldB(brow, 0, hi);
  v16bf a1 = ldA(arow, 32, hi), b1 = ldB(brow, 32, hi);
  v16bf a2, b2, a3, b3;
#pragma unroll 1
  for (int k0 = 0; k0 + 128 < K; k0 += 128) {
    a2 = ldA(arow, k0 + 64, hi);  b2 = ldB(brow, k0 + 64, hi);
    acc = wmma_bf16(a0, b0, acc);
    a3 = ldA(arow, k0 + 96, hi);  b3 = ldB(brow, k0 + 96, hi);
    acc = wmma_bf16(a1, b1, acc);
    a0 = ldA(arow, k0 + 128, hi); b0 = ldB(brow, k0 + 128, hi);
    acc = wmma_bf16(a2, b2, acc);
    a1 = ldA(arow, k0 + 160, hi); b1 = ldB(brow, k0 + 160, hi);
    acc = wmma_bf16(a3, b3, acc);
  }
  a2 = ldA(arow, K - 64, hi); b2 = ldB(brow, K - 64, hi);
  a3 = ldA(arow, K - 32, hi); b3 = ldB(brow, K - 32, hi);
  acc = wmma_bf16(a0, b0, acc);
  acc = wmma_bf16(a1, b1, acc);
  acc = wmma_bf16(a2, b2, acc);
  acc = wmma_bf16(a3, b3, acc);
}

template <int K>
__device__ __forceinline__ void gemm_pipe2(const unsigned short* arow,
                                           const unsigned short* b0row,
                                           const unsigned short* b1row, int hi,
                                           v8f& acc0, v8f& acc1) {
  static_assert(K % 128 == 0 && K >= 256, "K layout");
  v16bf a0 = ldA(arow, 0, hi),  p0 = ldB(b0row, 0, hi),  q0 = ldB(b1row, 0, hi);
  v16bf a1 = ldA(arow, 32, hi), p1 = ldB(b0row, 32, hi), q1 = ldB(b1row, 32, hi);
  v16bf a2, p2, q2, a3, p3, q3;
#pragma unroll 1
  for (int k0 = 0; k0 + 128 < K; k0 += 128) {
    a2 = ldA(arow, k0 + 64, hi);
    p2 = ldB(b0row, k0 + 64, hi);  q2 = ldB(b1row, k0 + 64, hi);
    acc0 = wmma_bf16(a0, p0, acc0);
    acc1 = wmma_bf16(a0, q0, acc1);
    a3 = ldA(arow, k0 + 96, hi);
    p3 = ldB(b0row, k0 + 96, hi);  q3 = ldB(b1row, k0 + 96, hi);
    acc0 = wmma_bf16(a1, p1, acc0);
    acc1 = wmma_bf16(a1, q1, acc1);
    a0 = ldA(arow, k0 + 128, hi);
    p0 = ldB(b0row, k0 + 128, hi); q0 = ldB(b1row, k0 + 128, hi);
    acc0 = wmma_bf16(a2, p2, acc0);
    acc1 = wmma_bf16(a2, q2, acc1);
    a1 = ldA(arow, k0 + 160, hi);
    p1 = ldB(b0row, k0 + 160, hi); q1 = ldB(b1row, k0 + 160, hi);
    acc0 = wmma_bf16(a3, p3, acc0);
    acc1 = wmma_bf16(a3, q3, acc1);
  }
  a2 = ldA(arow, K - 64, hi);
  p2 = ldB(b0row, K - 64, hi); q2 = ldB(b1row, K - 64, hi);
  a3 = ldA(arow, K - 32, hi);
  p3 = ldB(b0row, K - 32, hi); q3 = ldB(b1row, K - 32, hi);
  acc0 = wmma_bf16(a0, p0, acc0);
  acc1 = wmma_bf16(a0, q0, acc1);
  acc0 = wmma_bf16(a1, p1, acc0);
  acc1 = wmma_bf16(a1, q1, acc1);
  acc0 = wmma_bf16(a2, p2, acc0);
  acc1 = wmma_bf16(a2, q2, acc1);
  acc0 = wmma_bf16(a3, p3, acc0);
  acc1 = wmma_bf16(a3, q3, acc1);
}

__device__ __forceinline__ float block_sum256(float v, float* red) {
  __syncthreads();
  for (int o = 16; o; o >>= 1) v += __shfl_xor(v, o, 32);
  if ((threadIdx.x & 31) == 0) red[threadIdx.x >> 5] = v;
  __syncthreads();
  if (threadIdx.x == 0) {
    float s = 0.f;
    for (int i = 0; i < 8; ++i) s += red[i];
    red[0] = s;
  }
  __syncthreads();
  return red[0];
}

// ---------------------------------------------------------------------------
// Workspace layout (bytes)
// ---------------------------------------------------------------------------
// sums: [0]=sum_mag [1]=sum_mag2 [2]=up_scalar [3]=sum_d [4]=sum_sig_mb
//       [5]=sum_gate [6]=sum_align
#define WS_CB      0                              // codebook bf16   (4096x1024)
#define WS_CBT     (WS_CB   + 8388608)            // codebook^T bf16 (1024x4096)
#define WS_TT      (WS_CBT  + 8388608)            // T^T bf16 (4096x4096)
#define WS_MT      (WS_TT   + 33554432)           // M^T bf16 (4096x4096)
#define WS_AT      (WS_MT   + 33554432)           // anchors^T bf16 (32x4096)
#define WS_WT      (WS_AT   + 262144)             // W^T bf16 (32x1024)
#define WS_CN      (WS_WT   + 65536)              // codebook row norms (4096 f32)
#define WS_SUMS    (WS_CN   + 16384)              // 8 floats
#define WS_TOTAL   (WS_SUMS + 64)

// ---------------------------------------------------------------------------
// Output layout (floats)
// ---------------------------------------------------------------------------
#define O_ZQ    ((size_t)0)                       // 2*ROWS*DIM (complex interleaved)
#define O_SOFT  (O_ZQ + (size_t)2 * ROWS * DIMM)  // ROWS*NSYM
#define O_HARD  (O_SOFT + (size_t)ROWS * NSYM)    // ROWS
#define O_DMEAN (O_HARD + ROWS)                   // 1
#define O_ENT   (O_DMEAN + 1)                     // ROWS
#define O_ALIGN (O_ENT + ROWS)                    // 1
#define O_TRANS (O_ALIGN + 1)                     // 1

// ---------------------------------------------------------------------------
// Prep kernels
// ---------------------------------------------------------------------------
__global__ void k_zero(float* sums) {
  if (threadIdx.x < 8) sums[threadIdx.x] = 0.0f;
}

__global__ void k_stats(const float* __restrict__ zr,
                        const float* __restrict__ zi, float* sums) {
  __shared__ float red[8];
  float s = 0.f, s2 = 0.f;
  for (size_t i = (size_t)blockIdx.x * blockDim.x + threadIdx.x; i < N_ELEM;
       i += (size_t)gridDim.x * blockDim.x) {
    float a = zr[i], b = zi[i];
    float m = sqrtf(a * a + b * b);
    s += m;
    s2 += m * m;
  }
  s = block_sum256(s, red);
  s2 = block_sum256(s2, red);
  if (threadIdx.x == 0) {
    atomicAdd(&sums[0], s);
    atomicAdd(&sums[1], s2);
  }
}

__global__ void k_up(float* sums) {
  if (threadIdx.x == 0) {
    double n = (double)N_ELEM;
    double s = (double)sums[0], s2 = (double)sums[1];
    double var = (s2 - s * s / n) / (n - 1.0);   // ddof=1
    double x = var / (1.0 + 1e-6);
    double sp = (x > 20.0) ? x : log1p(exp(x));  // softplus
    sums[2] = (float)sp;
  }
}

// codebook f32 -> bf16 (row major), plus per-row squared norms
__global__ void k_cbconv(const float* __restrict__ cb, unsigned short* cb_bf,
                         float* cn) {
  __shared__ float red[8];
  int n = blockIdx.x;
  float s = 0.f;
  for (int c = threadIdx.x; c < TWOD; c += 256) {
    float v = cb[(size_t)n * TWOD + c];
    cb_bf[(size_t)n * TWOD + c] = f2bf(v);
    s += v * v;
  }
  s = block_sum256(s, red);
  if (threadIdx.x == 0) cn[n] = s;
}

// in f32 (R x C) -> out bf16 (C x R) transposed, 32x32 LDS tiles
__global__ void k_transpose_bf16(const float* __restrict__ in,
                                 unsigned short* out, int R, int C) {
  __shared__ float tile[32][33];
  int bc = blockIdx.x * 32, br = blockIdx.y * 32;
  int tx = threadIdx.x & 31, ty = threadIdx.x >> 5;  // 32x8 threads
  for (int i = ty; i < 32; i += 8) {
    int r = br + i, c = bc + tx;
    tile[i][tx] = (r < R && c < C) ? in[(size_t)r * C + c] : 0.0f;
  }
  __syncthreads();
  for (int i = ty; i < 32; i += 8) {
    int c = bc + i, r = br + tx;
    if (c < C && r < R) out[(size_t)c * R + r] = f2bf(tile[tx][i]);
  }
}

// ---------------------------------------------------------------------------
// Main fused kernel: one workgroup per 16-row tile of (b,s)
// ---------------------------------------------------------------------------
#define SMEM_A_FLAT_U16 (16 * TWOD)   // 16384 ushorts
#define SMEM_A_P_U16    (16 * NSYM)   // 65536 ushorts
#define SMEM_MAIN_BYTES (SMEM_A_FLAT_U16 * 2 + SMEM_A_P_U16 * 2 + \
                         (16 + 128 + 16 + 16 + 512 + 512 + 64) * 4)

__launch_bounds__(256)
__global__ void k_main(const float* __restrict__ z_real,
                       const float* __restrict__ z_imag,
                       const float* __restrict__ psd,
                       const float* __restrict__ prev_ent,
                       const float* __restrict__ anchor_b,
                       const unsigned short* __restrict__ cb_bf,
                       const unsigned short* __restrict__ cbT_bf,
                       const unsigned short* __restrict__ Tt,
                       const unsigned short* __restrict__ Mt,
                       const unsigned short* __restrict__ Wt,
                       const unsigned short* __restrict__ At,
                       const float* __restrict__ cn,
                       float* out_zq, float* out_soft, float* out_hard,
                       float* out_entropy, float* sums) {
  extern __shared__ unsigned char smem[];
  unsigned short* aFlat = (unsigned short*)smem;              // 16 x 1024 bf16
  unsigned short* aP    = aFlat + SMEM_A_FLAT_U16;            // 16 x 4096 bf16
  float* rn      = (float*)(aP + SMEM_A_P_U16);               // 16
  float* rowMaxW = rn + 16;                                   // 8 x 16
  float* rowMax  = rowMaxW + 128;                             // 16
  float* rowSum  = rowMax + 16;                               // 16
  float* projL   = rowSum + 16;                               // 16 x 32
  float* anchL   = projL + 512;                               // 16 x 32
  float* red     = anchL + 512;                               // 64

  const int tid = threadIdx.x;
  const int w = tid >> 5, lane = tid & 31;
  const int laneN = lane & 15, hi = lane >> 4;
  const int rowg0 = blockIdx.x * 16;

  // ---- init small LDS ----
  if (tid < 16) rn[tid] = 0.0f;
  if (tid < 128) rowMaxW[tid] = -1e30f;
  __syncthreads();

  // ---- stage A tiles into LDS as bf16; accumulate flat row norms ----
  for (int i = tid; i < 16 * TWOD; i += 256) {
    int r = i >> 10, c = i & (TWOD - 1);
    int rg = rowg0 + r;
    float v = (c < DIMM) ? z_real[(size_t)rg * DIMM + c]
                         : z_imag[(size_t)rg * DIMM + (c - DIMM)];
    aFlat[r * TWOD + c] = f2bf(v);
    atomicAdd(&rn[r], v * v);
  }
  for (int i = tid; i < 16 * NSYM; i += 256) {
    int r = i >> 12, c = i & (NSYM - 1);
    aP[r * NSYM + c] = f2bf(psd[(size_t)(rowg0 + r) * NSYM + c]);
  }
  __syncthreads();

  // ---- pass 1: d, graph_bias, meta_bias; logits -> global; row max ----
  const float up = sums[2];
  const float c1 = 0.01f * up;  // L2 * up_scalar
  float sum_d_loc = 0.f, sum_smb_loc = 0.f;
  const unsigned short* aFrow = aFlat + laneN * TWOD;
  const unsigned short* aProw = aP + laneN * NSYM;

#pragma unroll 1
  for (int t = 0; t < 32; ++t) {
    const int n0 = t * 128 + w * 16;
    const int colC = n0 + laneN;

    const unsigned short* brow = cb_bf + (size_t)colC * TWOD;
    const unsigned short* trow = Tt + (size_t)colC * NSYM;
    const unsigned short* mrow = Mt + (size_t)colC * NSYM;

    // Warm near cache for the next tile's B rows (global_prefetch_b8).
    if (t < 31) {
      __builtin_prefetch(brow + (size_t)128 * TWOD, 0, 1);
      __builtin_prefetch(trow + (size_t)128 * NSYM, 0, 1);
      __builtin_prefetch(mrow + (size_t)128 * NSYM, 0, 1);
    }

    // d accumulation (K = 1024), pipelined
    v8f acc = {};
    gemm_pipe1<TWOD>(aFrow, brow, hi, acc);

    // graph + meta bias (K = 4096), shared A fragment, pipelined
    v8f accg = {}, accm = {};
    gemm_pipe2<NSYM>(aProw, trow, mrow, hi, accg, accm);

    const float cnv = cn[colC];
    float logit[8];
    for (int r = 0; r < 8; ++r) {
      int m = hi * 8 + r;
      float dv = rn[m] + cnv - 2.0f * acc[r];
      sum_d_loc += dv;
      sum_smb_loc += sigmoidf(accm[r]);
      float lg = -dv + c1 * sigmoidf(accg[r]);
      logit[r] = lg;
      out_soft[(size_t)(rowg0 + m) * NSYM + colC] = lg;  // scratch: logits
    }
    // per-row max across the 16 lanes of each half
    for (int r = 0; r < 8; ++r) {
      float v = logit[r];
      for (int o = 8; o >= 1; o >>= 1) v = fmaxf(v, __shfl_xor(v, o, 32));
      if (laneN == 0) {
        float* dst = &rowMaxW[w * 16 + hi * 8 + r];
        *dst = fmaxf(*dst, v);
      }
    }
  }

  float bs = block_sum256(sum_d_loc, red);
  float bs2 = block_sum256(sum_smb_loc, red);
  if (tid == 0) {
    atomicAdd(&sums[3], bs);
    atomicAdd(&sums[4], bs2);
  }
  __threadfence();
  __syncthreads();
  if (tid < 16) {
    float mx = -1e30f;
    for (int ww = 0; ww < 8; ++ww) mx = fmaxf(mx, rowMaxW[ww * 16 + tid]);
    rowMax[tid] = mx;
  }
  __syncthreads();

  // ---- pass 2a: sum of exp per row; stash exp() as bf16 A-matrix in aP ----
  for (int mi = 0; mi < 2; ++mi) {
    int m = w * 2 + mi;
    float rmax = rowMax[m];
    size_t base = (size_t)(rowg0 + m) * NSYM;
    float s = 0.f;
    for (int c = lane; c < NSYM; c += 32) {
      float e = __expf(out_soft[base + c] - rmax);
      s += e;
      aP[m * NSYM + c] = f2bf(e);
    }
    for (int o = 16; o; o >>= 1) s += __shfl_xor(s, o, 32);
    if (lane == 0) rowSum[m] = s;
  }
  __syncthreads();

  // ---- pass 2b: probs out, entropy, argmax, gate ----
  for (int mi = 0; mi < 2; ++mi) {
    int m = w * 2 + mi;
    float inv = 1.0f / rowSum[m];
    size_t base = (size_t)(rowg0 + m) * NSYM;
    float ent = 0.f, bmax = -1e30f;
    int bidx = 0;
    for (int c = lane; c < NSYM; c += 32) {
      float e = bf2f(aP[m * NSYM + c]);
      float p = e * inv;
      out_soft[base + c] = p;
      ent += -p * __logf(p + 1e-9f);
      if (e > bmax) { bmax = e; bidx = c; }
    }
    for (int o = 16; o; o >>= 1) ent += __shfl_xor(ent, o, 32);
    for (int o = 16; o; o >>= 1) {
      float om = __shfl_xor(bmax, o, 32);
      int oi = __shfl_xor(bidx, o, 32);
      if (om > bmax || (om == bmax && oi < bidx)) { bmax = om; bidx = oi; }
    }
    if (lane == 0) {
      int rg = rowg0 + m;
      out_entropy[rg] = ent;
      out_hard[rg] = (float)bidx;
      float dH = fabsf(ent - prev_ent[rg]);
      float gate = sigmoidf((dH - 0.5f) * 10.0f);
      atomicAdd(&sums[5], gate);
    }
  }
  __syncthreads();

  // ---- pass 3: z_q = (exp-vals @ codebook) / rowSum  (K = 4096) ----
  const unsigned short* aErow = aP + laneN * NSYM;
#pragma unroll 1
  for (int j = 0; j < 8; ++j) {
    int t = j * 8 + w;               // 64 column tiles over 1024 dims
    int dcol = t * 16 + laneN;
    v8f acc = {};
    const unsigned short* brow = cbT_bf + (size_t)dcol * NSYM;
    gemm_pipe1<NSYM>(aErow, brow, hi, acc);
    for (int r = 0; r < 8; ++r) {
      int m = hi * 8 + r;
      float v = acc[r] / rowSum[m];
      // complex64 interleaved: real at 2*k, imag at 2*k+1
      size_t zb = ((size_t)(rowg0 + m) * DIMM + (dcol & (DIMM - 1))) * 2 +
                  (dcol >> 9);
      out_zq[zb] = v;
    }
  }

  // ---- pass 4: proj = flat@W + b (waves 0,1); anchor = exp@A (waves 2,3) ----
  if (w < 2) {
    int colP = w * 16 + laneN;
    v8f acc = {};
    const unsigned short* brow = Wt + (size_t)colP * TWOD;
    gemm_pipe1<TWOD>(aFrow, brow, hi, acc);
    float bias = anchor_b[colP];
    for (int r = 0; r < 8; ++r) projL[(hi * 8 + r) * 32 + colP] = acc[r] + bias;
  } else if (w < 4) {
    int colP = (w - 2) * 16 + laneN;
    v8f acc = {};
    const unsigned short* brow = At + (size_t)colP * NSYM;
    gemm_pipe1<NSYM>(aErow, brow, hi, acc);
    for (int r = 0; r < 8; ++r) anchL[(hi * 8 + r) * 32 + colP] = acc[r];
  }
  __syncthreads();

  // ---- pass 5: alignment partial sum ----
  float al = 0.f;
  for (int i = tid; i < 512; i += 256) {
    int m = i >> 5, p = i & 31;
    float a = anchL[i] / rowSum[m];
    float df = projL[i] - a;
    al += df * df;
  }
  al = block_sum256(al, red);
  if (tid == 0) atomicAdd(&sums[6], al);
}

// ---------------------------------------------------------------------------
// Finalize scalar outputs
// ---------------------------------------------------------------------------
__global__ void k_finalize(const float* sums, float* out) {
  if (threadIdx.x == 0) {
    double nd = (double)ROWS * NSYM;
    double dmean = (double)sums[3] / nd;
    double smb = (double)sums[4] / nd;
    double gmean = (double)sums[5] / (double)ROWS;
    double align = (double)sums[6] / ((double)ROWS * PROJ);
    out[O_DMEAN] = (float)dmean;
    out[O_ALIGN] = (float)align;
    out[O_TRANS] = (float)(gmean * (dmean - smb));
  }
}

// ---------------------------------------------------------------------------
// Host-side launcher
// ---------------------------------------------------------------------------
extern "C" void kernel_launch(void* const* d_in, const int* in_sizes, int n_in,
                              void* d_out, int out_size, void* d_ws,
                              size_t ws_size, hipStream_t stream) {
  const float* z_real   = (const float*)d_in[0];
  const float* z_imag   = (const float*)d_in[1];
  const float* psd      = (const float*)d_in[2];
  const float* prev_ent = (const float*)d_in[3];
  const float* codebook = (const float*)d_in[4];
  const float* tmat     = (const float*)d_in[5];
  const float* anchors  = (const float*)d_in[6];
  const float* anchorW  = (const float*)d_in[7];
  const float* anchorB  = (const float*)d_in[8];
  const float* meta     = (const float*)d_in[9];

  unsigned char* ws = (unsigned char*)d_ws;
  unsigned short* cb_bf  = (unsigned short*)(ws + WS_CB);
  unsigned short* cbT_bf = (unsigned short*)(ws + WS_CBT);
  unsigned short* Tt     = (unsigned short*)(ws + WS_TT);
  unsigned short* Mt     = (unsigned short*)(ws + WS_MT);
  unsigned short* At     = (unsigned short*)(ws + WS_AT);
  unsigned short* Wt     = (unsigned short*)(ws + WS_WT);
  float* cn   = (float*)(ws + WS_CN);
  float* sums = (float*)(ws + WS_SUMS);

  float* out = (float*)d_out;

  hipLaunchKernelGGL(k_zero, dim3(1), dim3(32), 0, stream, sums);
  hipLaunchKernelGGL(k_stats, dim3(2048), dim3(256), 0, stream, z_real, z_imag,
                     sums);
  hipLaunchKernelGGL(k_up, dim3(1), dim3(32), 0, stream, sums);
  hipLaunchKernelGGL(k_cbconv, dim3(NSYM), dim3(256), 0, stream, codebook,
                     cb_bf, cn);
  // transposed bf16 copies (in R x C -> out C x R)
  hipLaunchKernelGGL(k_transpose_bf16, dim3(TWOD / 32, NSYM / 32), dim3(256), 0,
                     stream, codebook, cbT_bf, NSYM, TWOD);
  hipLaunchKernelGGL(k_transpose_bf16, dim3(NSYM / 32, NSYM / 32), dim3(256), 0,
                     stream, tmat, Tt, NSYM, NSYM);
  hipLaunchKernelGGL(k_transpose_bf16, dim3(NSYM / 32, NSYM / 32), dim3(256), 0,
                     stream, meta, Mt, NSYM, NSYM);
  hipLaunchKernelGGL(k_transpose_bf16, dim3(1, NSYM / 32), dim3(256), 0, stream,
                     anchors, At, NSYM, PROJ);
  hipLaunchKernelGGL(k_transpose_bf16, dim3(1, TWOD / 32), dim3(256), 0, stream,
                     anchorW, Wt, TWOD, PROJ);

  float* out_zq   = out + O_ZQ;
  float* out_soft = out + O_SOFT;
  float* out_hard = out + O_HARD;
  float* out_ent  = out + O_ENT;

  hipLaunchKernelGGL(k_main, dim3(ROWS / 16), dim3(256), SMEM_MAIN_BYTES,
                     stream, z_real, z_imag, psd, prev_ent, anchorB, cb_bf,
                     cbT_bf, Tt, Mt, Wt, At, cn, out_zq, out_soft, out_hard,
                     out_ent, sums);

  hipLaunchKernelGGL(k_finalize, dim3(1), dim3(32), 0, stream, sums, out);
}
